// NormalNet_20985210208673
// MI455X (gfx1250) — matrix operand order repaired and checked by reading, not compiled, not run on hardware
//
#include <hip/hip_runtime.h>
#include <hip/hip_bf16.h>

typedef __attribute__((ext_vector_type(16))) __bf16 v16bf;
typedef __attribute__((ext_vector_type(8)))  __bf16 v8bf;
typedef __attribute__((ext_vector_type(8)))  float  v8f;

#define BN_EPS 1e-5f
#define SLOPE  0.01f

__device__ __forceinline__ float lrelu(float x) { return x > 0.f ? x : SLOPE * x; }

// ---------------- utility kernels ----------------
__global__ void k_zero(float* __restrict__ p, int n) {
  int i = blockIdx.x * blockDim.x + threadIdx.x;
  if (i < n) p[i] = 0.f;
}

__global__ void k_degree(const int* __restrict__ dst, float* __restrict__ deg, int E, int Nn) {
  int e = blockIdx.x * blockDim.x + threadIdx.x;
  if (e >= E) return;
  int d = dst[e];
  if ((unsigned)d < (unsigned)Nn) atomicAdd(&deg[d], 1.0f);
}

__global__ void k_dinv(float* __restrict__ deg, int Nn) {
  int i = blockIdx.x * blockDim.x + threadIdx.x;
  if (i < Nn) deg[i] = rsqrtf(deg[i] + 1.0f);
}

__global__ void k_enorm(const int* __restrict__ src, const int* __restrict__ dst,
                        const float* __restrict__ dinv, float* __restrict__ enorm,
                        int E, int Nn) {
  int e = blockIdx.x * blockDim.x + threadIdx.x;
  if (e >= E) return;
  int s = src[e], d = dst[e];
  float w = 0.f;
  if ((unsigned)s < (unsigned)Nn && (unsigned)d < (unsigned)Nn) w = dinv[s] * dinv[d];
  enorm[e] = w;
}

__global__ void k_selfnorm(const float* __restrict__ dinv, float* __restrict__ sn, int Nn) {
  int i = blockIdx.x * blockDim.x + threadIdx.x;
  if (i < Nn) sn[i] = dinv[i] * dinv[i];
}

// pack f32 [Nn,C] -> bf16 [Nn,Kp] zero-padded in K
__global__ void k_pack_x(const float* __restrict__ x, __bf16* __restrict__ xbf,
                         int Nn, int C, int Kp) {
  int idx = blockIdx.x * blockDim.x + threadIdx.x;
  int total = Nn * Kp;
  if (idx >= total) return;
  int r = idx / Kp, k = idx - r * Kp;
  float v = (k < C) ? x[(size_t)r * C + k] : 0.f;
  xbf[idx] = (__bf16)v;
}

// pack W f32 [K,Cout] -> Wt bf16 [Cout,Kp] (transposed, K-padded)
__global__ void k_pack_wt(const float* __restrict__ W, __bf16* __restrict__ Wt,
                          int K, int Cout, int Kp) {
  int idx = blockIdx.x * blockDim.x + threadIdx.x;
  int total = Cout * Kp;
  if (idx >= total) return;
  int co = idx / Kp, k = idx - co * Kp;
  float v = (k < K) ? W[(size_t)k * Cout + co] : 0.f;
  Wt[idx] = (__bf16)v;
}

// ---------------- WMMA GEMM: C[M,Nc] = A[M,Kp](bf16) x Bt[Nc,Kp]^T(bf16) ----------------
__global__ void k_gemm_bf16(const __bf16* __restrict__ Abf, const __bf16* __restrict__ Bt,
                            float* __restrict__ C, int Mtiles, int Kp, int Nc) {
  int wave  = threadIdx.x >> 5;
  int lane  = threadIdx.x & 31;
  int mtile = blockIdx.x * 4 + wave;
  if (mtile >= Mtiles) return;          // wave-uniform guard; EXEC stays all-ones inside
  int ntile = blockIdx.y;
  int sub   = lane & 15;
  int half  = lane >> 4;

  // A fragment (16x32 bf16): lane(sub,half) holds K = 8*half+[0..8) and 8*half+16+[0..8)
  const __bf16* ap = Abf + (size_t)(mtile * 16 + sub) * Kp + half * 8;
  // B fragment (32x16 bf16) from transposed weights: lane holds K = 8*half+[0..8) and 8*half+16+[0..8), col=sub
  const __bf16* bp = Bt + (size_t)(ntile * 16 + sub) * Kp + half * 8;

  v8f acc = {};
  for (int k0 = 0; k0 < Kp; k0 += 32) {
    v8bf a_lo = *(const v8bf*)(ap + k0);
    v8bf a_hi = *(const v8bf*)(ap + k0 + 16);
    v8bf b_lo = *(const v8bf*)(bp + k0);
    v8bf b_hi = *(const v8bf*)(bp + k0 + 16);
    v16bf a = __builtin_shufflevector(a_lo, a_hi, 0,1,2,3,4,5,6,7,8,9,10,11,12,13,14,15);
    v16bf b = __builtin_shufflevector(b_lo, b_hi, 0,1,2,3,4,5,6,7,8,9,10,11,12,13,14,15);
    acc = __builtin_amdgcn_wmma_f32_16x16x32_bf16(false, a, false, b, (short)0, acc, false, false);
  }
  // D layout: VGPR v -> row m = 8*half + v, col n = sub
  float* cp = C + (size_t)(mtile * 16 + half * 8) * Nc + ntile * 16 + sub;
#pragma unroll
  for (int v = 0; v < 8; ++v) cp[(size_t)v * Nc] = acc[v];
}

// agg = xw * self_norm[row] + bias[col]
__global__ void k_self_bias(const float* __restrict__ xw, const float* __restrict__ sn,
                            const float* __restrict__ bias, float* __restrict__ agg,
                            int Nn, int C) {
  int idx = blockIdx.x * blockDim.x + threadIdx.x;
  int total = Nn * C;
  if (idx >= total) return;
  int r = idx / C, c = idx - r * C;
  agg[idx] = xw[idx] * sn[r] + bias[c];
}

// agg[dst] += xw[src] * enorm[e]   (4 channels per thread, 128-bit gather)
__global__ void k_scatter(const float* __restrict__ xw, float* __restrict__ agg,
                          const int* __restrict__ src, const int* __restrict__ dst,
                          const float* __restrict__ enorm, int E, int C, int Nn) {
  int e = blockIdx.x * blockDim.x + threadIdx.x;
  if (e >= E) return;
  int s = src[e], d = dst[e];
  if ((unsigned)s >= (unsigned)Nn || (unsigned)d >= (unsigned)Nn) return;
  int c0 = blockIdx.y * 4;
  float w = enorm[e];
  const float4 v = *(const float4*)(xw + (size_t)s * C + c0);
  float* ap = agg + (size_t)d * C + c0;
  atomicAdd(ap + 0, v.x * w);
  atomicAdd(ap + 1, v.y * w);
  atomicAdd(ap + 2, v.z * w);
  atomicAdd(ap + 3, v.w * w);
}

// per-block LDS column partials -> global atomics; stats[0..512)=sum, [512..1024)=sumsq
__global__ void k_bn_stats(const float* __restrict__ agg, float* __restrict__ stats,
                           int Nn, int C, int rowsPerBlock) {
  extern __shared__ float sm[];            // 2*C floats
  float* ss = sm;
  float* sq = sm + C;
  for (int c = threadIdx.x; c < C; c += blockDim.x) { ss[c] = 0.f; sq[c] = 0.f; }
  __syncthreads();
  int r0 = blockIdx.x * rowsPerBlock;
  int r1 = r0 + rowsPerBlock; if (r1 > Nn) r1 = Nn;
  for (int r = r0; r < r1; ++r) {
    const float* row = agg + (size_t)r * C;
    for (int c = threadIdx.x; c < C; c += blockDim.x) {
      float v = row[c];
      ss[c] += v;                          // each channel owned by one thread
      sq[c] += v * v;
    }
  }
  __syncthreads();
  for (int c = threadIdx.x; c < C; c += blockDim.x) {
    atomicAdd(&stats[c], ss[c]);
    atomicAdd(&stats[512 + c], sq[c]);
  }
}

__global__ void k_bn_finalize(float* __restrict__ stats, float invN, int C) {
  int c = blockIdx.x * blockDim.x + threadIdx.x;
  if (c >= C) return;
  float mean = stats[c] * invN;
  float var  = stats[512 + c] * invN - mean * mean;
  stats[1024 + c] = mean;
  stats[1536 + c] = rsqrtf(var + BN_EPS);
}

// x = lrelu(gamma*(agg-mean)*invstd+beta); write f32 in place + bf16 K-padded repack
__global__ void k_bn_apply_pack(float* __restrict__ agg, __bf16* __restrict__ xbf,
                                const float* __restrict__ stats,
                                const float* __restrict__ gamma, const float* __restrict__ beta,
                                int Nn, int C, int Kp) {
  int idx = blockIdx.x * blockDim.x + threadIdx.x;
  int total = Nn * Kp;
  if (idx >= total) return;
  int r = idx / Kp, k = idx - r * Kp;
  float y = 0.f;
  if (k < C) {
    float v = agg[(size_t)r * C + k];
    y = lrelu(gamma[k] * (v - stats[1024 + k]) * stats[1536 + k] + beta[k]);
    agg[(size_t)r * C + k] = y;
  }
  xbf[idx] = (__bf16)y;
}

// head: lrelu(x@W1+b1) -> tanh(.@W2+b2) -> L2 normalize rows
__global__ void k_head(const float* __restrict__ x, const float* __restrict__ W1,
                       const float* __restrict__ b1, const float* __restrict__ W2,
                       const float* __restrict__ b2, float* __restrict__ out, int Nn) {
  __shared__ float sW1[32 * 16], sW2[16 * 3], sb1[16], sb2[3];
  for (int i = threadIdx.x; i < 512; i += blockDim.x) sW1[i] = W1[i];
  for (int i = threadIdx.x; i < 48;  i += blockDim.x) sW2[i] = W2[i];
  if (threadIdx.x < 16) sb1[threadIdx.x] = b1[threadIdx.x];
  if (threadIdx.x < 3)  sb2[threadIdx.x] = b2[threadIdx.x];
  __syncthreads();
  int r = blockIdx.x * blockDim.x + threadIdx.x;
  if (r >= Nn) return;
  const float* xr = x + (size_t)r * 32;
  float h1[16];
#pragma unroll
  for (int j = 0; j < 16; ++j) {
    float acc = sb1[j];
    for (int k = 0; k < 32; ++k) acc += xr[k] * sW1[k * 16 + j];
    h1[j] = lrelu(acc);
  }
  float h2[3], nrm = 0.f;
#pragma unroll
  for (int j = 0; j < 3; ++j) {
    float acc = sb2[j];
    for (int k = 0; k < 16; ++k) acc += h1[k] * sW2[k * 3 + j];
    acc = tanhf(acc);
    h2[j] = acc;
    nrm += acc * acc;
  }
  float inv = 1.0f / (sqrtf(nrm) + 1e-12f);
  out[(size_t)r * 3 + 0] = h2[0] * inv;
  out[(size_t)r * 3 + 1] = h2[1] * inv;
  out[(size_t)r * 3 + 2] = h2[2] * inv;
}

// ---------------- host launch ----------------
static inline int cdiv(long long a, long long b) { return (int)((a + b - 1) / b); }

extern "C" void kernel_launch(void* const* d_in, const int* in_sizes, int n_in,
                              void* d_out, int out_size, void* d_ws, size_t ws_size,
                              hipStream_t stream) {
  (void)n_in; (void)out_size; (void)ws_size;
  static const int Hh[15] = {7, 32, 64, 128, 256, 256, 512, 512, 256, 256, 128, 64, 32, 16, 3};

  const float* z2  = (const float*)d_in[0];
  const int*   ei  = (const int*)d_in[1];
  const int Nn = in_sizes[0] / 7;
  const int E  = in_sizes[1] / 2;
  const int* src = ei;
  const int* dst = ei + E;

  // workspace carve-up (256B aligned)
  size_t off = 0;
  auto carve = [&](size_t bytes) {
    void* p = (char*)d_ws + off;
    off += (bytes + 255) & ~(size_t)255;
    return p;
  };
  float*  bufA  = (float*)carve((size_t)Nn * 512 * 4);   // agg / x (f32)
  float*  bufB  = (float*)carve((size_t)Nn * 512 * 4);   // xw (f32)
  __bf16* xbf   = (__bf16*)carve((size_t)Nn * 512 * 2);  // packed activations
  __bf16* wbf   = (__bf16*)carve((size_t)512 * 512 * 2); // packed W^T
  float*  dinv  = (float*)carve((size_t)Nn * 4);         // deg -> dinv in place
  float*  enorm = (float*)carve((size_t)E * 4);
  float*  selfn = (float*)carve((size_t)Nn * 4);
  float*  stats = (float*)carve((size_t)2048 * 4);       // sum|sumsq|mean|invstd

  const int TB = 256;

  // graph normalization terms (loop-invariant)
  k_zero<<<cdiv(Nn, TB), TB, 0, stream>>>(dinv, Nn);
  k_degree<<<cdiv(E, TB), TB, 0, stream>>>(dst, dinv, E, Nn);
  k_dinv<<<cdiv(Nn, TB), TB, 0, stream>>>(dinv, Nn);
  k_enorm<<<cdiv(E, TB), TB, 0, stream>>>(src, dst, dinv, enorm, E, Nn);
  k_selfnorm<<<cdiv(Nn, TB), TB, 0, stream>>>(dinv, selfn, Nn);

  // layer 0 input: pack z2 [N,7] -> bf16 [N,32]
  k_pack_x<<<cdiv((long long)Nn * 32, TB), TB, 0, stream>>>(z2, xbf, Nn, 7, 32);

  const int Mtiles = cdiv(Nn, 16);
  for (int i = 0; i < 12; ++i) {
    const int Cin  = Hh[i];
    const int Cout = Hh[i + 1];
    const int Kp   = (Cin + 31) & ~31;
    const int KpN  = (Cout + 31) & ~31;
    const float* W     = (const float*)d_in[2 + 4 * i + 0];
    const float* bias  = (const float*)d_in[2 + 4 * i + 1];
    const float* gamma = (const float*)d_in[2 + 4 * i + 2];
    const float* beta  = (const float*)d_in[2 + 4 * i + 3];

    k_pack_wt<<<cdiv((long long)Cout * Kp, TB), TB, 0, stream>>>(W, wbf, Cin, Cout, Kp);

    dim3 gg(cdiv(Mtiles, 4), Cout / 16);
    k_gemm_bf16<<<gg, 128, 0, stream>>>(xbf, wbf, bufB, Mtiles, Kp, Cout);

    k_self_bias<<<cdiv((long long)Nn * Cout, TB), TB, 0, stream>>>(bufB, selfn, bias, bufA, Nn, Cout);

    dim3 gs(cdiv(E, TB), Cout / 4);
    k_scatter<<<gs, TB, 0, stream>>>(bufB, bufA, src, dst, enorm, E, Cout, Nn);

    k_zero<<<cdiv(1024, TB), TB, 0, stream>>>(stats, 1024);
    const int NB = 512;
    k_bn_stats<<<NB, TB, 2 * Cout * sizeof(float), stream>>>(bufA, stats, Nn, Cout, cdiv(Nn, NB));
    k_bn_finalize<<<cdiv(Cout, TB), TB, 0, stream>>>(stats, 1.0f / (float)Nn, Cout);

    k_bn_apply_pack<<<cdiv((long long)Nn * KpN, TB), TB, 0, stream>>>(
        bufA, xbf, stats, gamma, beta, Nn, Cout, KpN);
  }

  const float* W1 = (const float*)d_in[50];
  const float* b1 = (const float*)d_in[51];
  const float* W2 = (const float*)d_in[52];
  const float* b2 = (const float*)d_in[53];
  k_head<<<cdiv(Nn, 128), 128, 0, stream>>>(bufA, W1, b1, W2, b2, (float*)d_out, Nn);
}
